// BitAllocationNetwork_50508815401405
// MI455X (gfx1250) — matrix operand order repaired and checked by reading, not compiled, not run on hardware
//
#include <hip/hip_runtime.h>
#include <hip/hip_bf16.h>
#include <math.h>

typedef __attribute__((ext_vector_type(16))) _Float16 v16h;
typedef __attribute__((ext_vector_type(8)))  float    v8f;
typedef __attribute__((ext_vector_type(4)))  float    f32x4;

#define NUM_GROUPS  8
#define GROUP_SIZE  512
#define FEATURE_DIM 4096
#define T_DIM       2048
#define B_DIM       16
#define H_DIM       16
#define SPLIT_T     8
#define ROWS_PER    (T_DIM / SPLIT_T)   // 256 rows per block

// d_out offsets (flat float32, tuple return order)
#define OFF_DISCRETE 0
#define OFF_GIDX     (B_DIM * NUM_GROUPS)                        // 128
#define OFF_EMB      (OFF_GIDX + B_DIM * FEATURE_DIM)            // 65664
#define OFF_BLOSS    (OFF_EMB + B_DIM * NUM_GROUPS * GROUP_SIZE) // 131200
#define OFF_DLOSS    (OFF_BLOSS + 1)                             // 131201
#define OFF_PROBS    (OFF_DLOSS + 1)                             // 131202

// ---------------------------------------------------------------------------
// Kernel A: streaming reduction of importance_scores -> per-(b,g,split) sums.
// 512 MB read once -> non-temporal b128 loads; ~22 us HBM floor.
// grid = (128, 8): x = b*8+g slab, y = T split. 256 threads = 8 wave32s.
// Each thread owns a fixed float4 column; pointer walks rows with a constant
// 32 KB stride. Two independent accumulators + unroll 4 => 8 b128 loads in
// flight per wave (4 KB) for HBM latency hiding.
// ---------------------------------------------------------------------------
__global__ void ban_reduce_kernel(const float* __restrict__ scores,
                                  float* __restrict__ partial) {
    const int bg    = blockIdx.x;          // 0..127
    const int split = blockIdx.y;          // 0..7
    const int b     = bg >> 3;
    const int g     = bg & 7;
    const float* base = scores + (size_t)b * T_DIM * FEATURE_DIM
                               + (size_t)g * GROUP_SIZE
                               + (size_t)split * ROWS_PER * FEATURE_DIM;
    const int tid = threadIdx.x;

    // 256 threads tile 2 rows x 128 float4; advance 4 rows per iteration
    // (two loads), 64 iterations covers 256 rows.
    const float* p = base + (size_t)(tid >> 7) * FEATURE_DIM
                          + (size_t)(tid & 127) * 4;

    f32x4 acc0 = (f32x4){0.f, 0.f, 0.f, 0.f};
    f32x4 acc1 = (f32x4){0.f, 0.f, 0.f, 0.f};
    #pragma unroll 4
    for (int it = 0; it < ROWS_PER / 4; ++it) {
        f32x4 v0 = __builtin_nontemporal_load((const f32x4*)p);
        f32x4 v1 = __builtin_nontemporal_load((const f32x4*)(p + 2 * FEATURE_DIM));
        acc0 += v0;
        acc1 += v1;
        p += 4 * FEATURE_DIM;
    }
    acc0 += acc1;
    float s = acc0.x + acc0.y + acc0.z + acc0.w;

    // wave32 reduction
    for (int off = 16; off > 0; off >>= 1) s += __shfl_down(s, off);

    __shared__ float red[8];
    if ((tid & 31) == 0) red[tid >> 5] = s;
    __syncthreads();
    if (tid == 0) {
        float t = 0.f;
        #pragma unroll
        for (int i = 0; i < 8; ++i) t += red[i];
        // fold the full mean factor here; summing partials stays correct
        partial[bg * SPLIT_T + split] = t * (1.0f / ((float)T_DIM * (float)GROUP_SIZE));
    }
}

// ---------------------------------------------------------------------------
// Kernel B: one wave32. group_imp -> WMMA MLP -> gumbel softmax -> allocation.
// Layer1: D = A(16x8 pad K->32, f16) x W1(8x16 pad, f16), f32 accum (v_wmma).
// Layer2: D = h(16x16 pad K->32)     x W2(16x8 pad N->16), f32 accum (v_wmma).
// ---------------------------------------------------------------------------
__global__ void ban_mlp_kernel(const float* __restrict__ partial,
                               const float* __restrict__ u,
                               const float* __restrict__ W1g,
                               const float* __restrict__ b1g,
                               const float* __restrict__ lngg,
                               const float* __restrict__ lnbg,
                               const float* __restrict__ W2g,
                               const float* __restrict__ b2g,
                               float* __restrict__ out,
                               float* __restrict__ bitw) {
    __shared__ float gimp[B_DIM * NUM_GROUPS];  // 128
    __shared__ float hbuf[B_DIM * H_DIM];       // 256
    __shared__ float logit[B_DIM * NUM_GROUPS]; // 128

    const int lane = threadIdx.x;               // 0..31
    const int col  = lane & 15;
    const bool hi  = (lane >= 16);

    // gather split partials -> group_imp
    for (int i = lane; i < B_DIM * NUM_GROUPS; i += 32) {
        float s = 0.f;
        #pragma unroll
        for (int k = 0; k < SPLIT_T; ++k) s += partial[i * SPLIT_T + k];
        gimp[i] = s;
    }
    __syncthreads();

    // ---- layer 1 fragments (A 16x32 f16 layout; lanes>=16 hold K=8..15 => 0)
    v16h a1; v16h bf1;
    #pragma unroll
    for (int j = 0; j < 16; ++j) { a1[j] = (_Float16)0.f; bf1[j] = (_Float16)0.f; }
    if (!hi) {
        #pragma unroll
        for (int j = 0; j < NUM_GROUPS; ++j) a1[j] = (_Float16)gimp[col * NUM_GROUPS + j];
        #pragma unroll
        for (int j = 0; j < NUM_GROUPS; ++j) bf1[j] = (_Float16)W1g[j * H_DIM + col];
    }
    v8f c1 = {};
    c1 = __builtin_amdgcn_wmma_f32_16x16x32_f16(false, a1, false, bf1,
                                                (short)0, c1, false, false);

    // bias + exact GELU + LayerNorm (row values live across 16 lanes per VGPR)
    const float bias1 = b1g[col];
    const float lg = lngg[col], lb = lnbg[col];
    #pragma unroll
    for (int r = 0; r < 8; ++r) {
        float x  = c1[r] + bias1;
        float gl = 0.5f * x * (1.0f + erff(x * 0.70710678118654752f));
        float s = gl;
        s += __shfl_xor(s, 1); s += __shfl_xor(s, 2);
        s += __shfl_xor(s, 4); s += __shfl_xor(s, 8);
        const float mu = s * (1.0f / 16.0f);
        const float d  = gl - mu;
        float v = d * d;
        v += __shfl_xor(v, 1); v += __shfl_xor(v, 2);
        v += __shfl_xor(v, 4); v += __shfl_xor(v, 8);
        const float var = v * (1.0f / 16.0f);
        const float hn  = d * rsqrtf(var + 1e-5f) * lg + lb;
        const int row = r + (hi ? 8 : 0);
        hbuf[row * H_DIM + col] = hn;
    }
    __syncthreads();

    // ---- layer 2 fragments: A = h (K=16), B = W2 (16x8, N padded to 16)
    v16h a2; v16h bf2;
    #pragma unroll
    for (int j = 0; j < 16; ++j) { a2[j] = (_Float16)0.f; bf2[j] = (_Float16)0.f; }
    if (!hi) {
        #pragma unroll
        for (int j = 0; j < H_DIM; ++j) a2[j] = (_Float16)hbuf[col * H_DIM + j];   // K=0..15
        if (col < NUM_GROUPS) {
            #pragma unroll
            for (int j = 0; j < H_DIM; ++j) bf2[j] = (_Float16)W2g[j * NUM_GROUPS + col];
        }
    }
    v8f c2 = {};
    c2 = __builtin_amdgcn_wmma_f32_16x16x32_f16(false, a2, false, bf2,
                                                (short)0, c2, false, false);

    const float bias2 = (col < NUM_GROUPS) ? b2g[col] : 0.f;
    #pragma unroll
    for (int r = 0; r < 8; ++r) {
        const int row = r + (hi ? 8 : 0);
        if (col < NUM_GROUPS) logit[row * NUM_GROUPS + col] = c2[r] + bias2;
    }
    __syncthreads();

    // ---- per-row postprocess: 16 lanes, one batch row each (tiny serial work)
    float sum_bits = 0.f, sum_ent = 0.f;
    if (lane < B_DIM) {
        const int b = lane;
        const float L[3] = {2.f, 4.f, 8.f};
        float lgt[NUM_GROUPS], p[NUM_GROUPS], alloc[NUM_GROUPS];
        float mx = -1e30f;
        #pragma unroll
        for (int g = 0; g < NUM_GROUPS; ++g) {
            const float uu  = u[b * NUM_GROUPS + g];
            const float gum = -logf(-logf(uu + 1e-8f) + 1e-8f);
            lgt[g] = logit[b * NUM_GROUPS + g] + gum;
            mx = fmaxf(mx, lgt[g]);
        }
        float Z = 0.f;
        #pragma unroll
        for (int g = 0; g < NUM_GROUPS; ++g) { p[g] = expf(lgt[g] - mx); Z += p[g]; }
        const float invZ = 1.0f / Z;
        float S = 0.f;
        #pragma unroll
        for (int g = 0; g < NUM_GROUPS; ++g) {
            p[g] *= invZ;
            out[OFF_PROBS + b * NUM_GROUPS + g] = p[g];
            sum_ent += -p[g] * logf(p[g] + 1e-8f);
            alloc[g] = 2.0f + p[g] * 6.0f;                 // MIN + p*(MAX-MIN)
            S += alloc[g];
        }
        const float scale = 32.0f / S;                      // budget = 4*8
        #pragma unroll
        for (int g = 0; g < NUM_GROUPS; ++g) {
            const float a = fminf(8.0f, fmaxf(2.0f, alloc[g] * scale));
            int   idx = 0;
            float bd  = fabsf(a - L[0]);
            #pragma unroll
            for (int l = 1; l < 3; ++l) {
                const float dd = fabsf(a - L[l]);
                if (dd < bd) { bd = dd; idx = l; }          // first-min like argmin
            }
            const float hard = L[idx];
            out[OFF_DISCRETE + b * NUM_GROUPS + g] = hard;  // ST value == hard
            sum_bits += hard;
            // bit_w = softmax(exp(-|hard - L|))
            float w[3], wsum = 0.f;
            #pragma unroll
            for (int l = 0; l < 3; ++l) { w[l] = expf(expf(-fabsf(hard - L[l]))); wsum += w[l]; }
            #pragma unroll
            for (int l = 0; l < 3; ++l)
                bitw[(b * NUM_GROUPS + g) * 3 + l] = w[l] / wsum;
        }
    }
    // losses: lanes >= 16 contribute zero
    for (int m = 1; m < 32; m <<= 1) {
        sum_bits += __shfl_xor(sum_bits, m);
        sum_ent  += __shfl_xor(sum_ent,  m);
    }
    if (lane == 0) {
        const float meanb = sum_bits * (1.0f / (B_DIM * NUM_GROUPS));
        out[OFF_BLOSS] = (meanb - 4.0f) * (meanb - 4.0f);
        out[OFF_DLOSS] = sum_ent * (1.0f / B_DIM);
    }
}

// ---------------------------------------------------------------------------
// Kernel C: embeddings = bit_w @ bit_embeddings, plus group_indices fill.
// ---------------------------------------------------------------------------
__global__ void ban_emb_kernel(const float* __restrict__ bitw,
                               const float* __restrict__ E,
                               float* __restrict__ out) {
    const int bg = blockIdx.x;            // 0..127
    const int b  = bg >> 3;
    const int g  = bg & 7;
    const float w0 = bitw[bg * 3 + 0];
    const float w1 = bitw[bg * 3 + 1];
    const float w2 = bitw[bg * 3 + 2];
    for (int d = threadIdx.x; d < GROUP_SIZE; d += blockDim.x) {
        const float e = w0 * E[d] + w1 * E[GROUP_SIZE + d] + w2 * E[2 * GROUP_SIZE + d];
        out[OFF_EMB  + bg * GROUP_SIZE + d] = e;
        out[OFF_GIDX + b * FEATURE_DIM + g * GROUP_SIZE + d] = (float)g;
    }
}

extern "C" void kernel_launch(void* const* d_in, const int* in_sizes, int n_in,
                              void* d_out, int out_size, void* d_ws, size_t ws_size,
                              hipStream_t stream) {
    const float* scores = (const float*)d_in[0];
    const float* u      = (const float*)d_in[1];
    const float* W1     = (const float*)d_in[2];
    const float* b1     = (const float*)d_in[3];
    const float* ln_g   = (const float*)d_in[4];
    const float* ln_b   = (const float*)d_in[5];
    const float* W2     = (const float*)d_in[6];
    const float* b2     = (const float*)d_in[7];
    const float* E      = (const float*)d_in[8];
    float* out = (float*)d_out;

    float* partial = (float*)d_ws;                 // 128*8 = 1024 floats
    float* bitw    = (float*)d_ws + 1024;          // 128*3 =  384 floats

    ban_reduce_kernel<<<dim3(B_DIM * NUM_GROUPS, SPLIT_T), 256, 0, stream>>>(scores, partial);
    ban_mlp_kernel<<<1, 32, 0, stream>>>(partial, u, W1, b1, ln_g, ln_b, W2, b2, out, bitw);
    ban_emb_kernel<<<B_DIM * NUM_GROUPS, 256, 0, stream>>>(bitw, E, out);
}
